// LocalizedFiltering_3332894622536
// MI455X (gfx1250) — compile-verified, matrix operand order checked by reading, not run on hardware
//
#include <hip/hip_runtime.h>
#include <hip/hip_bf16.h>

// ---------------------------------------------------------------------------
// LocalizedFiltering fused pipeline for MI455X (gfx1250, wave32, WMMA)
//   pre:    convert inputs/caches to bf16; convert+transpose W1/W2 into
//           B-matrix-ready bf16 layout Bt[n][k]
//   stage1: [x(t-1);x(t)]_bf16 @ Bt1 -> out1 bf16           (M=8192,N=2048,K=8192)
//   stage2: [o1(t-1);o1(t)]_bf16 @ Bt2 + b2 + residual -> y (M=8192,N=4096,K=4096)
//   stage3: in-place RMSNorm(y) * ln_weight
// GEMM: block tile 128x256, 8 waves (2M x 4N), 64x64 wave tile = 4x4 WMMA
// frags, double-buffered LDS (one barrier per K-step), bf16 ops / f32 accum.
// ---------------------------------------------------------------------------

#define B_  4
#define T_  2048
#define D_  4096
#define H_  2048
#define M_  (B_ * T_)      // 8192 token rows
#define LOG2T 11           // T_ == 2048

#define LDA   40           // LDS row pitch (ushorts); rows 80B -> 16B aligned
#define ABUF  (128 * LDA)  // one A buffer (ushorts)
#define BBUF  (256 * LDA)  // one B buffer (ushorts)

// workspace layout (ushort elements, all 16B aligned)
#define OFF_XBF  0u                                   // [M_, D_]   bf16 inputs
#define OFF_LF1  (OFF_XBF + (unsigned)M_ * D_)        // [B_, D_]
#define OFF_LF2  (OFF_LF1 + (unsigned)B_ * D_)        // [B_, H_]
#define OFF_BT1  (OFF_LF2 + (unsigned)B_ * H_)        // [H_, 2*D_]  Bt1[n][k]
#define OFF_BT2  (OFF_BT1 + (unsigned)H_ * 2 * D_)    // [D_, 2*H_]  Bt2[n][k]
#define OFF_O1   (OFF_BT2 + (unsigned)D_ * 2 * H_)    // [M_, H_]    out1 bf16

typedef __attribute__((ext_vector_type(16))) __bf16 v16bf;
typedef __attribute__((ext_vector_type(8)))  float  v8f;

union BFrag { int   i[8]; v16bf v; };
union FAcc  { float f[8]; v8f  v; };

__device__ __forceinline__ unsigned short f2bf(float f) {
  unsigned u = __float_as_uint(f);
  u += 0x7FFFu + ((u >> 16) & 1u);        // round-to-nearest-even
  return (unsigned short)(u >> 16);
}
__device__ __forceinline__ unsigned pack2(float lo, float hi) {
  return (unsigned)f2bf(lo) | ((unsigned)f2bf(hi) << 16);
}
// K index covered by fragment VGPR v, half hi (ISA 16-bit A/B layout, wave32)
__device__ __forceinline__ int frag_kpair(int v, int hi) {
  return ((v < 4) ? (v * 2) : (16 + (v - 4) * 2)) + hi * 8;
}

// ---------------------------------------------------------------------------
// Pre-pass: f32 -> bf16 elementwise (count multiple of 2048; 8 elems/thread)
// ---------------------------------------------------------------------------
__global__ __launch_bounds__(256)
void cvt_f32_bf16(const float* __restrict__ src, unsigned short* __restrict__ dst) {
  int base = (blockIdx.x * 256 + threadIdx.x) * 8;
  float4 a = *(const float4*)(src + base);
  float4 b = *(const float4*)(src + base + 4);
  uint4 o;
  o.x = pack2(a.x, a.y);
  o.y = pack2(a.z, a.w);
  o.z = pack2(b.x, b.y);
  o.w = pack2(b.z, b.w);
  *(uint4*)(dst + base) = o;
}

// ---------------------------------------------------------------------------
// Pre-pass: transpose+convert block: dst_bf16[c][dstKOff + r] = src[r][srcColOff+c]
// 32x32 tile via LDS, 256 threads, grid (C/32, R/32)
// ---------------------------------------------------------------------------
__global__ __launch_bounds__(256)
void transpose_cvt(const float* __restrict__ src, int ldS, int srcColOff,
                   unsigned short* __restrict__ dst, int ldD, int dstKOff) {
  __shared__ float t[32][33];
  const int tx = threadIdx.x & 31, ty = threadIdx.x >> 5;  // 32 x 8
  const int rbase = blockIdx.y * 32, cbase = blockIdx.x * 32;
#pragma unroll
  for (int i = 0; i < 4; ++i)
    t[ty + i * 8][tx] = src[(size_t)(rbase + ty + i * 8) * ldS + srcColOff + cbase + tx];
  __syncthreads();
#pragma unroll
  for (int i = 0; i < 4; ++i)
    dst[(size_t)(cbase + ty + i * 8) * ldD + dstKOff + rbase + tx] = f2bf(t[tx][ty + i * 8]);
}

// ---------------------------------------------------------------------------
// One K-phase of the 128x256 block GEMM (bf16 operands, f32 accumulate).
// Per thread: 2 A rows (r0, r0+64) and 4 B rows (r0+64j), column chunk c8.
// Double-buffered LDS; one barrier per K-step.
// ---------------------------------------------------------------------------
template <int KPH>
__device__ __forceinline__ void gemm_phase(
    const unsigned short* __restrict__ pA0, const unsigned short* __restrict__ pA1,
    const unsigned short* __restrict__ pB0, const unsigned short* __restrict__ pB1,
    const unsigned short* __restrict__ pB2, const unsigned short* __restrict__ pB3,
    unsigned short* As, unsigned short* Bs,
    int r0, int c8, int waveM, int waveN, int hi, int l16, FAcc (&acc)[4][4]) {
  const int aw0 = r0 * LDA + c8;
  const int aw1 = (r0 + 64) * LDA + c8;

  auto compute = [&](const unsigned short* A_r, const unsigned short* B_r) {
    BFrag af[4], bf[4];
#pragma unroll
    for (int mi = 0; mi < 4; ++mi) {
      int rowl = waveM * 64 + mi * 16 + l16;
#pragma unroll
      for (int v = 0; v < 8; ++v)
        af[mi].i[v] = *(const int*)&A_r[rowl * LDA + frag_kpair(v, hi)];
    }
#pragma unroll
    for (int ni = 0; ni < 4; ++ni) {
      int coll = waveN * 64 + ni * 16 + l16;
#pragma unroll
      for (int v = 0; v < 8; ++v)
        bf[ni].i[v] = *(const int*)&B_r[coll * LDA + frag_kpair(v, hi)];
    }
#pragma unroll
    for (int mi = 0; mi < 4; ++mi)
#pragma unroll
      for (int ni = 0; ni < 4; ++ni)
        acc[mi][ni].v = __builtin_amdgcn_wmma_f32_16x16x32_bf16(
            false, af[mi].v, false, bf[ni].v, (short)0, acc[mi][ni].v,
            false, false);
  };

  // prologue: stage tile k0=0 into buffer 0
  uint4 ra0 = *(const uint4*)(pA0);
  uint4 ra1 = *(const uint4*)(pA1);
  uint4 rb0 = *(const uint4*)(pB0);
  uint4 rb1 = *(const uint4*)(pB1);
  uint4 rb2 = *(const uint4*)(pB2);
  uint4 rb3 = *(const uint4*)(pB3);
  *(uint4*)&As[aw0] = ra0;
  *(uint4*)&As[aw1] = ra1;
  *(uint4*)&Bs[aw0] = rb0;
  *(uint4*)&Bs[aw1] = rb1;
  *(uint4*)&Bs[(r0 + 128) * LDA + c8] = rb2;
  *(uint4*)&Bs[(r0 + 192) * LDA + c8] = rb3;
  __syncthreads();

  int buf = 0;
  for (int k0 = 0; k0 < KPH - 32; k0 += 32) {
    // fetch next tile (global), streamed A gets an explicit prefetch ahead
    ra0 = *(const uint4*)(pA0 + k0 + 32);
    ra1 = *(const uint4*)(pA1 + k0 + 32);
    rb0 = *(const uint4*)(pB0 + k0 + 32);
    rb1 = *(const uint4*)(pB1 + k0 + 32);
    rb2 = *(const uint4*)(pB2 + k0 + 32);
    rb3 = *(const uint4*)(pB3 + k0 + 32);
    __builtin_prefetch(pA0 + k0 + 64, 0, 1);
    __builtin_prefetch(pA1 + k0 + 64, 0, 1);

    // compute on current buffer
    compute(As + buf * ABUF, Bs + buf * BBUF);

    // stage next tile into the other buffer
    unsigned short* A_w = As + (buf ^ 1) * ABUF;
    unsigned short* B_w = Bs + (buf ^ 1) * BBUF;
    *(uint4*)&A_w[aw0] = ra0;
    *(uint4*)&A_w[aw1] = ra1;
    *(uint4*)&B_w[aw0] = rb0;
    *(uint4*)&B_w[aw1] = rb1;
    *(uint4*)&B_w[(r0 + 128) * LDA + c8] = rb2;
    *(uint4*)&B_w[(r0 + 192) * LDA + c8] = rb3;
    __syncthreads();
    buf ^= 1;
  }
  compute(As + buf * ABUF, Bs + buf * BBUF);
  __syncthreads();
}

// ---------------------------------------------------------------------------
// Stage 1: out1 = [x(t-1);x(t)] @ Bt1 + b1    (bf16 out)
// grid (H/256, M/128)
// ---------------------------------------------------------------------------
__global__ __launch_bounds__(256)
void lf_conv1_gemm(const unsigned short* __restrict__ xbf,
                   const unsigned short* __restrict__ lf1bf,
                   const unsigned short* __restrict__ Bt1,
                   const float* __restrict__ b1,
                   unsigned short* __restrict__ out1) {
  __shared__ __align__(16) unsigned short As[2 * ABUF];
  __shared__ __align__(16) unsigned short Bs[2 * BBUF];

  const int tid = threadIdx.x;
  const int lane = tid & 31, wave = tid >> 5;
  const int waveM = wave & 1, waveN = wave >> 1;   // 2M x 4N waves
  const int hi = lane >> 4, l16 = lane & 15;
  const int m0 = blockIdx.y * 128, n0 = blockIdx.x * 256;

  const int r0 = tid >> 2;             // tile row 0..63
  const int c8 = (tid & 3) * 8;        // column chunk (8 bf16)

  const int row0 = m0 + r0, row1 = m0 + r0 + 64;
  const int t0 = row0 & (T_ - 1), b0i = row0 >> LOG2T;
  const int t1 = row1 & (T_ - 1), b1i = row1 >> LOG2T;
  const unsigned short* pA0p =
      ((t0 == 0) ? lf1bf + b0i * D_ : xbf + (size_t)(row0 - 1) * D_) + c8;
  const unsigned short* pA1p =
      ((t1 == 0) ? lf1bf + b1i * D_ : xbf + (size_t)(row1 - 1) * D_) + c8;
  const unsigned short* pA0c = xbf + (size_t)row0 * D_ + c8;
  const unsigned short* pA1c = xbf + (size_t)row1 * D_ + c8;
  const unsigned short* pB0 = Bt1 + (size_t)(n0 + r0) * (2 * D_) + c8;
  const unsigned short* pB1 = pB0 + (size_t)64 * (2 * D_);
  const unsigned short* pB2 = pB0 + (size_t)128 * (2 * D_);
  const unsigned short* pB3 = pB0 + (size_t)192 * (2 * D_);

  FAcc acc[4][4];
#pragma unroll
  for (int mi = 0; mi < 4; ++mi)
#pragma unroll
    for (int ni = 0; ni < 4; ++ni)
#pragma unroll
      for (int r = 0; r < 8; ++r) acc[mi][ni].f[r] = 0.0f;

  gemm_phase<D_>(pA0p, pA1p, pB0, pB1, pB2, pB3, As, Bs,
                 r0, c8, waveM, waveN, hi, l16, acc);
  gemm_phase<D_>(pA0c, pA1c, pB0 + D_, pB1 + D_, pB2 + D_, pB3 + D_, As, Bs,
                 r0, c8, waveM, waveN, hi, l16, acc);

#pragma unroll
  for (int mi = 0; mi < 4; ++mi)
#pragma unroll
    for (int ni = 0; ni < 4; ++ni)
#pragma unroll
      for (int r = 0; r < 8; ++r) {
        int mg = m0 + waveM * 64 + mi * 16 + r + 8 * hi;
        int ng = n0 + waveN * 64 + ni * 16 + l16;
        out1[(size_t)mg * H_ + ng] = f2bf(acc[mi][ni].f[r] + b1[ng]);
      }
}

// ---------------------------------------------------------------------------
// Stage 2: y = [o1(t-1);o1(t)] @ Bt2 + b2 + residual   (f32 out, pre-norm)
// grid (D/256, M/128)
// ---------------------------------------------------------------------------
__global__ __launch_bounds__(256)
void lf_conv2_gemm(const unsigned short* __restrict__ out1,
                   const unsigned short* __restrict__ lf2bf,
                   const unsigned short* __restrict__ Bt2,
                   const float* __restrict__ b2,
                   const float* __restrict__ resid,
                   float* __restrict__ y) {
  __shared__ __align__(16) unsigned short As[2 * ABUF];
  __shared__ __align__(16) unsigned short Bs[2 * BBUF];

  const int tid = threadIdx.x;
  const int lane = tid & 31, wave = tid >> 5;
  const int waveM = wave & 1, waveN = wave >> 1;
  const int hi = lane >> 4, l16 = lane & 15;
  const int m0 = blockIdx.y * 128, n0 = blockIdx.x * 256;

  const int r0 = tid >> 2;
  const int c8 = (tid & 3) * 8;

  const int row0 = m0 + r0, row1 = m0 + r0 + 64;
  const int t0 = row0 & (T_ - 1), b0i = row0 >> LOG2T;
  const int t1 = row1 & (T_ - 1), b1i = row1 >> LOG2T;
  const unsigned short* pA0p =
      ((t0 == 0) ? lf2bf + b0i * H_ : out1 + (size_t)(row0 - 1) * H_) + c8;
  const unsigned short* pA1p =
      ((t1 == 0) ? lf2bf + b1i * H_ : out1 + (size_t)(row1 - 1) * H_) + c8;
  const unsigned short* pA0c = out1 + (size_t)row0 * H_ + c8;
  const unsigned short* pA1c = out1 + (size_t)row1 * H_ + c8;
  const unsigned short* pB0 = Bt2 + (size_t)(n0 + r0) * (2 * H_) + c8;
  const unsigned short* pB1 = pB0 + (size_t)64 * (2 * H_);
  const unsigned short* pB2 = pB0 + (size_t)128 * (2 * H_);
  const unsigned short* pB3 = pB0 + (size_t)192 * (2 * H_);

  FAcc acc[4][4];
#pragma unroll
  for (int mi = 0; mi < 4; ++mi)
#pragma unroll
    for (int ni = 0; ni < 4; ++ni)
#pragma unroll
      for (int r = 0; r < 8; ++r) acc[mi][ni].f[r] = 0.0f;

  gemm_phase<H_>(pA0p, pA1p, pB0, pB1, pB2, pB3, As, Bs,
                 r0, c8, waveM, waveN, hi, l16, acc);
  gemm_phase<H_>(pA0c, pA1c, pB0 + H_, pB1 + H_, pB2 + H_, pB3 + H_, As, Bs,
                 r0, c8, waveM, waveN, hi, l16, acc);

#pragma unroll
  for (int mi = 0; mi < 4; ++mi)
#pragma unroll
    for (int ni = 0; ni < 4; ++ni)
#pragma unroll
      for (int r = 0; r < 8; ++r) {
        int mg = m0 + waveM * 64 + mi * 16 + r + 8 * hi;
        int ng = n0 + waveN * 64 + ni * 16 + l16;
        y[(size_t)mg * D_ + ng] =
            acc[mi][ni].f[r] + b2[ng] + resid[(size_t)mg * D_ + ng];
      }
}

// ---------------------------------------------------------------------------
// Stage 3: in-place row RMSNorm, one block (256 threads) per row of 4096
// ---------------------------------------------------------------------------
__global__ __launch_bounds__(256)
void lf_rmsnorm(float* __restrict__ y, const float* __restrict__ lnw) {
  __shared__ float red[256];
  float* p = y + (size_t)blockIdx.x * D_;
  const int tid = threadIdx.x;

  float v[16];
  float s = 0.0f;
#pragma unroll
  for (int j = 0; j < 16; ++j) {
    v[j] = p[tid + j * 256];
    s += v[j] * v[j];
  }
  red[tid] = s;
  __syncthreads();
  for (int off = 128; off > 0; off >>= 1) {
    if (tid < off) red[tid] += red[tid + off];
    __syncthreads();
  }
  float inv = rsqrtf(red[0] * (1.0f / (float)D_) + 1e-6f);
#pragma unroll
  for (int j = 0; j < 16; ++j) {
    int c = tid + j * 256;
    p[c] = v[j] * inv * lnw[c];
  }
}

// ---------------------------------------------------------------------------
extern "C" void kernel_launch(void* const* d_in, const int* in_sizes, int n_in,
                              void* d_out, int out_size, void* d_ws,
                              size_t ws_size, hipStream_t stream) {
  const float* x   = (const float*)d_in[0];  // inputs        [B,T,D]
  const float* lf1 = (const float*)d_in[1];  // lf1_cache     [B,1,D]
  const float* lf2 = (const float*)d_in[2];  // lf2_cache     [B,1,H]
  const float* W1  = (const float*)d_in[3];  // conv1_weight  [D,D]
  const float* W2  = (const float*)d_in[4];  // conv2_weight  [H,2D]
  const float* b1  = (const float*)d_in[5];  // conv1_bias    [H]
  const float* b2  = (const float*)d_in[6];  // conv2_bias    [D]
  const float* lnw = (const float*)d_in[7];  // ln_weight     [D]
  float* out = (float*)d_out;                // [B*T, D] f32

  unsigned short* ws = (unsigned short*)d_ws;
  unsigned short* xbf   = ws + OFF_XBF;
  unsigned short* lf1bf = ws + OFF_LF1;
  unsigned short* lf2bf = ws + OFF_LF2;
  unsigned short* Bt1   = ws + OFF_BT1;
  unsigned short* Bt2   = ws + OFF_BT2;
  unsigned short* o1    = ws + OFF_O1;

  dim3 blk(256);
  // pre-pass conversions (one-time, bandwidth-trivial vs 550 GFLOP of GEMM)
  cvt_f32_bf16<<<dim3((M_ * D_) / 2048), blk, 0, stream>>>(x, xbf);
  cvt_f32_bf16<<<dim3((B_ * D_) / 2048), blk, 0, stream>>>(lf1, lf1bf);
  cvt_f32_bf16<<<dim3((B_ * H_) / 2048), blk, 0, stream>>>(lf2, lf2bf);
  // Bt1[n][k] : k<D -> W1[k][n] ; k>=D -> W1[k-D][H+n]        (n<H, k<2D)
  transpose_cvt<<<dim3(H_ / 32, D_ / 32), blk, 0, stream>>>(W1, D_, 0,  Bt1, 2 * D_, 0);
  transpose_cvt<<<dim3(H_ / 32, D_ / 32), blk, 0, stream>>>(W1, D_, H_, Bt1, 2 * D_, D_);
  // Bt2[n][k] : k<H -> W2[k][n] ; k>=H -> W2[k-H][D+n]        (n<D, k<2H)
  transpose_cvt<<<dim3(D_ / 32, H_ / 32), blk, 0, stream>>>(W2, 2 * D_, 0,  Bt2, 2 * H_, 0);
  transpose_cvt<<<dim3(D_ / 32, H_ / 32), blk, 0, stream>>>(W2, 2 * D_, D_, Bt2, 2 * H_, H_);

  lf_conv1_gemm<<<dim3(H_ / 256, M_ / 128), blk, 0, stream>>>(xbf, lf1bf, Bt1, b1, o1);
  lf_conv2_gemm<<<dim3(D_ / 256, M_ / 128), blk, 0, stream>>>(o1, lf2bf, Bt2, b2, x, out);
  lf_rmsnorm<<<dim3(M_), blk, 0, stream>>>(out, lnw);
}
